// MultiHeadInfoQuantizer_8048768713194
// MI455X (gfx1250) — matrix operand order repaired and checked by reading, not compiled, not run on hardware
//
#include <hip/hip_runtime.h>
#include <hip/hip_bf16.h>
#include <stdint.h>

// ---------------- CDNA5 (gfx1250) wave32 WMMA types ----------------
typedef __attribute__((ext_vector_type(16))) __bf16 v16bf;
typedef __attribute__((ext_vector_type(8)))  float  v8f;

#define N_TOK   16384   // B*T
#define INCH    512
#define CHN     2048
#define DDIM    256
#define MCODES  1024
#define LN_EPS  1e-5f

__device__ __forceinline__ unsigned short f2bf(float f) {
  unsigned int u = __float_as_uint(f);
  unsigned int r = u + 0x7FFFu + ((u >> 16) & 1u);
  return (unsigned short)(r >> 16);
}

__device__ __forceinline__ v16bf load_frag16(const uint32_t* p) {
  return *(const v16bf*)p;   // 32B aligned (tile*1024 + lane*32)
}

__device__ __forceinline__ void store_frag16(uint32_t* p, const uint32_t dw[8]) {
  uint4 a; a.x = dw[0]; a.y = dw[1]; a.z = dw[2]; a.w = dw[3];
  uint4 b; b.x = dw[4]; b.y = dw[5]; b.z = dw[6]; b.w = dw[7];
  ((uint4*)p)[0] = a;
  ((uint4*)p)[1] = b;
}

// ---------------- fragment packers (f32 rowmajor -> bf16 WMMA frags) ------
// A-matrix 16x32 bf16 layout (ISA 7.12.2): lane m=lane&15, half=lane>>4;
// vgpr v holds K pair k0 = (v>>2)*16 + half*8 + (v&3)*2.
__global__ void pack_a_frag_f32(const float* __restrict__ src, uint32_t* __restrict__ dst,
                                int Kt, int nTiles) {
  int tile = blockIdx.x * 8 + (threadIdx.x >> 5);
  if (tile >= nTiles) return;
  int lane = threadIdx.x & 31;
  int rt = tile / Kt, kt = tile % Kt;
  int half = lane >> 4, m = lane & 15;
  int K = Kt * 32;
  const float* row = src + (size_t)(rt * 16 + m) * K + kt * 32;
  uint32_t dw[8];
#pragma unroll
  for (int v = 0; v < 8; ++v) {
    int k0 = ((v >> 2) << 4) + (half << 3) + ((v & 3) << 1);
    dw[v] = (uint32_t)f2bf(row[k0]) | ((uint32_t)f2bf(row[k0 + 1]) << 16);
  }
  store_frag16(dst + ((size_t)tile * 32 + lane) * 8, dw);
}

// B-matrix 32x16 bf16 layout: lane n=lane&15, half=lane>>4; vgpr v holds
// K pair k0 = (v>>2)*16 + half*8 + (v&3)*2 for column n.
__global__ void pack_b_frag_f32(const float* __restrict__ src, uint32_t* __restrict__ dst,
                                int Ct, int nTiles) {
  int tile = blockIdx.x * 8 + (threadIdx.x >> 5);
  if (tile >= nTiles) return;
  int lane = threadIdx.x & 31;
  int kt = tile / Ct, ct = tile % Ct;
  int half = lane >> 4, n = lane & 15;
  int C = Ct * 16;
  const float* col = src + (size_t)(kt * 32) * C + ct * 16 + n;
  uint32_t dw[8];
#pragma unroll
  for (int v = 0; v < 8; ++v) {
    int k0 = ((v >> 2) << 4) + (half << 3) + ((v & 3) << 1);
    dw[v] = (uint32_t)f2bf(col[(size_t)k0 * C]) |
            ((uint32_t)f2bf(col[(size_t)(k0 + 1) * C]) << 16);
  }
  store_frag16(dst + ((size_t)tile * 32 + lane) * 8, dw);
}

// B = log(embedding)^T : B[k][n] = log(src[n*K + k]);  src is [C=M, K=D].
__global__ void pack_bT_log_frag(const float* __restrict__ src, uint32_t* __restrict__ dst,
                                 int Ct, int K, int nTiles) {
  int tile = blockIdx.x * 8 + (threadIdx.x >> 5);
  if (tile >= nTiles) return;
  int lane = threadIdx.x & 31;
  int kt = tile / Ct, ct = tile % Ct;
  int half = lane >> 4, n = lane & 15;
  const float* rowp = src + (size_t)(ct * 16 + n) * K + kt * 32;
  uint32_t dw[8];
#pragma unroll
  for (int v = 0; v < 8; ++v) {
    int k0 = ((v >> 2) << 4) + (half << 3) + ((v & 3) << 1);
    dw[v] = (uint32_t)f2bf(logf(rowp[k0])) |
            ((uint32_t)f2bf(logf(rowp[k0 + 1])) << 16);
  }
  store_frag16(dst + ((size_t)tile * 32 + lane) * 8, dw);
}

// ---------------- GEMM1: h = relu(LN(x@W1)) -> hfrag (A-layout bf16) ------
// Block: 16 rows x 2048 cols; 8 waves x 16 coltiles; K=512 (16 ktiles).
__global__ __launch_bounds__(256)
void gemm1_ln_relu(const uint32_t* __restrict__ xfrag, const uint32_t* __restrict__ w1frag,
                   const float* __restrict__ ln_g, const float* __restrict__ ln_b,
                   uint32_t* __restrict__ hfrag) {
  __shared__ unsigned short shh[8][16][256];  // per-wave 16x256 bf16 tile (64KB)
  __shared__ float red1[16][8];
  __shared__ float red2[16][8];
  const int rt = blockIdx.x;
  const int w = threadIdx.x >> 5;
  const int lane = threadIdx.x & 31;
  const int half = lane >> 4, ln = lane & 15;

  v8f acc[16];
#pragma unroll
  for (int t = 0; t < 16; ++t) acc[t] = (v8f)0.f;

  for (int kt = 0; kt < 16; ++kt) {
    const uint32_t* ap = xfrag + (((size_t)rt * 16 + kt) * 32 + lane) * 8;
    if (kt < 15) __builtin_prefetch(ap + 256 * 8, 0, 0);   // global_prefetch A tile 8 ktiles ahead
    v16bf a = load_frag16(ap);
#pragma unroll
    for (int t = 0; t < 16; ++t) {
      int ct = w * 16 + t;
      v16bf b = load_frag16(w1frag + (((size_t)kt * 128 + ct) * 32 + lane) * 8);
      acc[t] = __builtin_amdgcn_wmma_f32_16x16x32_bf16(false, a, false, b,
                                                       (short)0, acc[t], false, false);
    }
  }

  // LayerNorm stats over full 2048 cols of 16 rows.
  // C/D layout: acc[t][r] -> row = r + 8*half, col = (w*16+t)*16 + ln
  float s1[8], s2[8];
#pragma unroll
  for (int r = 0; r < 8; ++r) { s1[r] = 0.f; s2[r] = 0.f; }
#pragma unroll
  for (int t = 0; t < 16; ++t)
#pragma unroll
    for (int r = 0; r < 8; ++r) { float v = acc[t][r]; s1[r] += v; s2[r] += v * v; }
#pragma unroll
  for (int off = 1; off < 16; off <<= 1)
#pragma unroll
    for (int r = 0; r < 8; ++r) {
      s1[r] += __shfl_xor(s1[r], off, 32);
      s2[r] += __shfl_xor(s2[r], off, 32);
    }
  if (ln == 0) {
#pragma unroll
    for (int r = 0; r < 8; ++r) {
      red1[r + 8 * half][w] = s1[r];
      red2[r + 8 * half][w] = s2[r];
    }
  }
  __syncthreads();
  float mu[8], rstd[8];
#pragma unroll
  for (int r = 0; r < 8; ++r) {
    int row = r + 8 * half;
    float t1 = 0.f, t2 = 0.f;
#pragma unroll
    for (int ww = 0; ww < 8; ++ww) { t1 += red1[row][ww]; t2 += red2[row][ww]; }
    float m = t1 * (1.0f / CHN);
    float v = t2 * (1.0f / CHN) - m * m;
    mu[r] = m;
    rstd[r] = rsqrtf(v + LN_EPS);
  }
#pragma unroll
  for (int t = 0; t < 16; ++t) {
    int colw = t * 16 + ln;
    int col = w * 256 + colw;
    float g = ln_g[col], bb = ln_b[col];
#pragma unroll
    for (int r = 0; r < 8; ++r) {
      int row = r + 8 * half;
      float v = (acc[t][r] - mu[r]) * rstd[r] * g + bb;
      shh[w][row][colw] = f2bf(fmaxf(v, 0.f));
    }
  }
  __syncthreads();
  // Repack: wave w owns cols [w*256, w*256+256) == hfrag ktiles w*8..w*8+7.
#pragma unroll
  for (int kt2 = 0; kt2 < 8; ++kt2) {
    int ktg = w * 8 + kt2;
    uint32_t dw[8];
#pragma unroll
    for (int v = 0; v < 8; ++v) {
      int k0 = kt2 * 32 + ((v >> 2) << 4) + (half << 3) + ((v & 3) << 1);
      dw[v] = (uint32_t)shh[w][ln][k0] | ((uint32_t)shh[w][ln][k0 + 1] << 16);
    }
    store_frag16(hfrag + (((size_t)rt * 64 + ktg) * 32 + lane) * 8, dw);
  }
}

// ---------------- GEMM2: z = h@W2+b2 ; p = per-head log_softmax -----------
// Block: 16 rows x 256 cols (full row); 8 waves x 2 coltiles; K=2048.
__global__ __launch_bounds__(256)
void gemm2_softmax(const uint32_t* __restrict__ hfrag, const uint32_t* __restrict__ w2frag,
                   const float* __restrict__ b2, float* __restrict__ z_out,
                   float* __restrict__ p_ws, uint32_t* __restrict__ exfrag) {
  __shared__ float sz[16][256];
  __shared__ unsigned short sex[16][256];
  __shared__ float smax[16][4];
  __shared__ float slog[16][4];
  const int rt = blockIdx.x;
  const int w = threadIdx.x >> 5;
  const int lane = threadIdx.x & 31;
  const int half = lane >> 4, ln = lane & 15;

  v8f acc[2] = {(v8f)0.f, (v8f)0.f};
  for (int kt = 0; kt < 64; ++kt) {
    const uint32_t* ap = hfrag + (((size_t)rt * 64 + kt) * 32 + lane) * 8;
    if (kt < 63) __builtin_prefetch(ap + 256 * 8, 0, 0);
    v16bf a = load_frag16(ap);
#pragma unroll
    for (int t = 0; t < 2; ++t) {
      int ct = w * 2 + t;
      v16bf b = load_frag16(w2frag + (((size_t)kt * 16 + ct) * 32 + lane) * 8);
      acc[t] = __builtin_amdgcn_wmma_f32_16x16x32_bf16(false, a, false, b,
                                                       (short)0, acc[t], false, false);
    }
  }
#pragma unroll
  for (int t = 0; t < 2; ++t) {
    int col = (w * 2 + t) * 16 + ln;
    float bias = b2[col];
#pragma unroll
    for (int r = 0; r < 8; ++r) {
      int row = r + 8 * half;
      float zv = acc[t][r] + bias;
      z_out[((size_t)rt * 16 + row) * DDIM + col] = zv;   // output #1
      sz[row][col] = zv;
    }
  }
  __syncthreads();
  if (threadIdx.x < 64) {          // one (row, head) per thread
    int row = threadIdx.x >> 2, head = threadIdx.x & 3;
    const float* rp = &sz[row][head * 64];
    float mx = rp[0];
    for (int j = 1; j < 64; ++j) mx = fmaxf(mx, rp[j]);
    float se = 0.f;
    for (int j = 0; j < 64; ++j) se += expf(rp[j] - mx);
    smax[row][head] = mx;
    slog[row][head] = logf(se);
  }
  __syncthreads();
#pragma unroll
  for (int i = 0; i < 16; ++i) {
    int f = threadIdx.x + i * 256;
    int row = f >> 8, col = f & 255, head = col >> 6;
    float pv = sz[row][col] - smax[row][head] - slog[row][head];
    p_ws[((size_t)rt * 16 + row) * DDIM + col] = pv;
    sex[row][col] = f2bf(expf(pv));
  }
  __syncthreads();
  {  // pack exp(p) as A-fragments; wave w handles ktile w (K=256 -> 8 ktiles)
    uint32_t dw[8];
#pragma unroll
    for (int v = 0; v < 8; ++v) {
      int k0 = w * 32 + ((v >> 2) << 4) + (half << 3) + ((v & 3) << 1);
      dw[v] = (uint32_t)sex[ln][k0] | ((uint32_t)sex[ln][k0 + 1] << 16);
    }
    store_frag16(exfrag + (((size_t)rt * 8 + w) * 32 + lane) * 8, dw);
  }
}

// ---------------- GEMM3: argmax_m( exp(p) . log(E_m) ) --------------------
// (argmin of KL divergence == argmax of dot; self-term is row-constant.)
// Block: 16 rows x 1024 codes; 8 waves x 8 coltiles; K=256.
__global__ __launch_bounds__(256)
void gemm3_argmax(const uint32_t* __restrict__ exfrag, const uint32_t* __restrict__ lefrag,
                  int* __restrict__ idx_ws) {
  __shared__ float candv[16][8];
  __shared__ int   candi[16][8];
  const int rt = blockIdx.x;
  const int w = threadIdx.x >> 5;
  const int lane = threadIdx.x & 31;
  const int half = lane >> 4, ln = lane & 15;

  v8f acc[8];
#pragma unroll
  for (int t = 0; t < 8; ++t) acc[t] = (v8f)0.f;
  for (int kt = 0; kt < 8; ++kt) {
    v16bf a = load_frag16(exfrag + (((size_t)rt * 8 + kt) * 32 + lane) * 8);
#pragma unroll
    for (int t = 0; t < 8; ++t) {
      int ct = w * 8 + t;
      v16bf b = load_frag16(lefrag + (((size_t)kt * 64 + ct) * 32 + lane) * 8);
      acc[t] = __builtin_amdgcn_wmma_f32_16x16x32_bf16(false, a, false, b,
                                                       (short)0, acc[t], false, false);
    }
  }
  float bv[8]; int bi[8];
#pragma unroll
  for (int r = 0; r < 8; ++r) { bv[r] = -3.4e38f; bi[r] = 0; }
#pragma unroll
  for (int t = 0; t < 8; ++t) {
    int col = (w * 8 + t) * 16 + ln;
#pragma unroll
    for (int r = 0; r < 8; ++r) {
      float v = acc[t][r];
      if (v > bv[r]) { bv[r] = v; bi[r] = col; }
    }
  }
#pragma unroll
  for (int off = 1; off < 16; off <<= 1)
#pragma unroll
    for (int r = 0; r < 8; ++r) {
      float ov = __shfl_xor(bv[r], off, 32);
      int   oi = __shfl_xor(bi[r], off, 32);
      if (ov > bv[r] || (ov == bv[r] && oi < bi[r])) { bv[r] = ov; bi[r] = oi; }
    }
  if (ln == 0) {
#pragma unroll
    for (int r = 0; r < 8; ++r) {
      candv[r + 8 * half][w] = bv[r];
      candi[r + 8 * half][w] = bi[r];
    }
  }
  __syncthreads();
  if (threadIdx.x < 16) {
    int row = threadIdx.x;
    float best = candv[row][0]; int besti = candi[row][0];
#pragma unroll
    for (int ww = 1; ww < 8; ++ww) {
      float v = candv[row][ww]; int ii = candi[row][ww];
      if (v > best || (v == best && ii < besti)) { best = v; besti = ii; }
    }
    idx_ws[rt * 16 + row] = besti;
  }
}

// ---------------- finalize: q_st gather + masked KL loss ------------------
__global__ __launch_bounds__(256)
void finalize(const int* __restrict__ idx_ws, const float* __restrict__ embedding,
              const float* __restrict__ p_ws, const float* __restrict__ masks,
              float* __restrict__ q_out, float* __restrict__ loss) {
  const int wv = threadIdx.x >> 5;
  const int lane = threadIdx.x & 31;
  const int n = blockIdx.x * 8 + wv;
  const int code = idx_ws[n];
  const float* e = embedding + (size_t)code * DDIM;
  const float* p = p_ws + (size_t)n * DDIM;
  float* q = q_out + (size_t)n * DDIM;
  float kl = 0.f;
#pragma unroll
  for (int j = 0; j < 8; ++j) {
    int d = lane + j * 32;
    float ev = e[d];
    float pv = p[d];
    q[d] = ev;                         // q_st value == quantized
    kl += expf(pv) * (pv - logf(ev));
  }
#pragma unroll
  for (int off = 1; off < 32; off <<= 1) kl += __shfl_xor(kl, off, 32);
  if (lane == 0) atomicAdd(loss, (0.25f / 16.0f) * kl * masks[n]);
}

__global__ void zero_loss(float* loss) {
  if (threadIdx.x == 0 && blockIdx.x == 0) loss[0] = 0.f;
}

// ---------------- launcher ------------------------------------------------
extern "C" void kernel_launch(void* const* d_in, const int* in_sizes, int n_in,
                              void* d_out, int out_size, void* d_ws, size_t ws_size,
                              hipStream_t stream) {
  const float* x     = (const float*)d_in[0];
  const float* masks = (const float*)d_in[1];
  const float* W1    = (const float*)d_in[2];
  const float* ln_g  = (const float*)d_in[3];
  const float* ln_b  = (const float*)d_in[4];
  const float* W2    = (const float*)d_in[5];
  const float* b2    = (const float*)d_in[6];
  const float* emb   = (const float*)d_in[7];

  float* out   = (float*)d_out;
  float* z_out = out;                                    // [N,256]
  float* q_out = out + (size_t)N_TOK * DDIM;             // [N,256]
  float* loss  = out + (size_t)2 * N_TOK * DDIM;         // [1]

  char* ws = (char*)d_ws;
  uint32_t* xfrag  = (uint32_t*)(ws + 0);                // 16 MB
  uint32_t* w1frag = (uint32_t*)(ws + 16777216);         //  2 MB
  uint32_t* hfrag  = (uint32_t*)(ws + 18874368);         // 64 MB
  uint32_t* w2frag = (uint32_t*)(ws + 85983232);         //  1 MB
  uint32_t* lefrag = (uint32_t*)(ws + 87031808);         // 0.5 MB
  float*    p_ws   = (float*)   (ws + 87556096);         // 16 MB
  uint32_t* exfrag = (uint32_t*)(ws + 104333312);        //  8 MB
  int*      idx_ws = (int*)     (ws + 112721920);        // 64 KB

  zero_loss<<<1, 64, 0, stream>>>(loss);
  pack_a_frag_f32 <<<2048, 256, 0, stream>>>(x,   xfrag,  16,      16384);
  pack_b_frag_f32 <<< 256, 256, 0, stream>>>(W1,  w1frag, 128,     2048);
  pack_b_frag_f32 <<< 128, 256, 0, stream>>>(W2,  w2frag, 16,      1024);
  pack_bT_log_frag<<<  64, 256, 0, stream>>>(emb, lefrag, 64, 256, 512);
  gemm1_ln_relu   <<<1024, 256, 0, stream>>>(xfrag, w1frag, ln_g, ln_b, hfrag);
  gemm2_softmax   <<<1024, 256, 0, stream>>>(hfrag, w2frag, b2, z_out, p_ws, exfrag);
  gemm3_argmax    <<<1024, 256, 0, stream>>>(exfrag, lefrag, idx_ws);
  finalize        <<<2048, 256, 0, stream>>>(idx_ws, emb, p_ws, masks, q_out, loss);
}
